// STCatAttention_85899345920233
// MI455X (gfx1250) — matrix operand order repaired and checked by reading, not compiled, not run on hardware
//
#include <hip/hip_runtime.h>

typedef __attribute__((ext_vector_type(16))) __bf16 v16bf;
typedef __attribute__((ext_vector_type(8)))  __bf16 v8bf;
typedef __attribute__((ext_vector_type(2)))  __bf16 v2bf;
typedef __attribute__((ext_vector_type(8)))  float  v8f;

#define BN_   828      // B*N = 4*207
#define Lv    96
#define Sv    96
#define Hv    8
#define Ev    64
#define Dv    64

// padded LDS row strides (elements): dword-stride % 8 == 4  -> conflict-free b128 operands
#define PA    72       // Q/K staging rows (64 data + 8 pad)   : 144B = 36 dwords
#define PQ    104      // QKbf & V^T rows (96 data + 8 pad)    : 208B = 52 dwords
#define LPQ   (Lv * PQ)          // QKbf per-head channel stride (elems)

// LDS carve (bytes)
#define OFF_QKBF  0                       // 8 * 96 * 104 * 2 = 159744
#define OFF_BUFA  159744                  // 96*72*2 = 13824 (Q staging; reused as V^T [64][104])
#define OFF_BUFB  173568                  // 96*72*2 = 13824 (K staging)
#define LDS_BYTES 187392

static __device__ __forceinline__ v16bf cat8(v8bf a, v8bf b) {
  return __builtin_shufflevector(a, b, 0,1,2,3,4,5,6,7,8,9,10,11,12,13,14,15);
}

__global__ __launch_bounds__(256) void stcat_attention_kernel(
    const float* __restrict__ q,    // [bn, L, H, E]
    const float* __restrict__ k,    // [bn, S, H, E]
    const float* __restrict__ v,    // [bn, S, H, D]
    const float* __restrict__ mask, // [L, S]
    const float* __restrict__ as,   // [bn, H, L, S]
    const float* __restrict__ W,    // [H, 2H] (uniform -> scalar loads)
    const float* __restrict__ bv,   // [H]
    float* __restrict__ qk_out,     // [bn, H, L, S]
    float* __restrict__ v_out)      // [bn, L, H, D]
{
  extern __shared__ __attribute__((aligned(16))) char smem[];
  __bf16* QKbf = (__bf16*)(smem + OFF_QKBF);
  __bf16* bufA = (__bf16*)(smem + OFF_BUFA);
  __bf16* bufB = (__bf16*)(smem + OFF_BUFB);

  const int t    = threadIdx.x;
  const int bn   = blockIdx.x;
  const int lane = t & 31;
  const int wave = t >> 5;
  const int c    = lane >> 4;   // lane half: K sub-block select / +8 row offset
  const int mn   = lane & 15;   // M (A rows) or N (B/C/D cols) within tile

  // ---------------- Phase A: QK = Q K^T for all 8 heads ----------------
  for (int h = 0; h < Hv; ++h) {
    __syncthreads();  // previous head's WMMA reads of bufA/bufB done
    for (int i = t; i < Lv * Ev; i += 256) {
      int r = i >> 6, e = i & 63;
      bufA[r * PA + e] = (__bf16)q[(((size_t)bn * Lv + r) * Hv + h) * Ev + e];
      bufB[r * PA + e] = (__bf16)k[(((size_t)bn * Sv + r) * Hv + h) * Ev + e];
    }
    __syncthreads();

    float* __restrict__ qkp = qk_out + ((size_t)bn * Hv + h) * (Lv * Sv);
    for (int tile = wave; tile < 36; tile += 8) {
      const int lt = (tile / 6) * 16, st = (tile % 6) * 16;
      v8f acc = {};
#pragma unroll
      for (int eb = 0; eb < Ev; eb += 32) {
        const __bf16* ar = &bufA[(lt + mn) * PA + eb + 8 * c];
        v16bf Am = cat8(*(const v8bf*)ar, *(const v8bf*)(ar + 16));
        const __bf16* br = &bufB[(st + mn) * PA + eb + 8 * c];
        v16bf Bm = cat8(*(const v8bf*)br, *(const v8bf*)(br + 16));
        acc = __builtin_amdgcn_wmma_f32_16x16x32_bf16(
            false, Am, false, Bm, (short)0, acc, false, false);
      }
#pragma unroll
      for (int r = 0; r < 8; ++r) {
        const int l = lt + r + 8 * c;
        const int s = st + mn;
        const float val = acc[r] + mask[l * Sv + s];
        qkp[l * Sv + s] = val;
        QKbf[h * LPQ + l * PQ + s] = (__bf16)val;
      }
    }
  }
  __syncthreads();

  // ---------------- Phase B1: 16->8 head mix, all heads at once ----------------
  // scores[h](l,s) = temp * (b[h] + sum_k W[h,k]*QK[k](l,s) + sum_k W[h,8+k]*AS[k](l,s))
  // overwrites QKbf in place; float2 over position pairs for packed-f32 math.
  {
    const float temp = 0.125f;  // 1/sqrt(E)
    const float2* __restrict__ asp = (const float2*)(as + (size_t)bn * Hv * Lv * Sv);
    for (int p = t; p < (Lv * Sv) / 2; p += 256) {
      const int l = p / 48, sp = p - l * 48, s = 2 * sp;
      float2 acch[Hv];
#pragma unroll
      for (int h = 0; h < Hv; ++h) { float b0 = bv[h]; acch[h] = make_float2(b0, b0); }
#pragma unroll
      for (int kk = 0; kk < Hv; ++kk) {
        const unsigned u = *(const unsigned*)&QKbf[kk * LPQ + l * PQ + s];
        const float c0 = __uint_as_float(u << 16);
        const float c1 = __uint_as_float(u & 0xffff0000u);
#pragma unroll
        for (int h = 0; h < Hv; ++h) {
          const float w = W[h * 16 + kk];
          acch[h].x += w * c0; acch[h].y += w * c1;
        }
      }
#pragma unroll
      for (int kk = 0; kk < Hv; ++kk) {
        const float2 a2 = asp[kk * ((Lv * Sv) / 2) + p];
#pragma unroll
        for (int h = 0; h < Hv; ++h) {
          const float w = W[h * 16 + 8 + kk];
          acch[h].x += w * a2.x; acch[h].y += w * a2.y;
        }
      }
#pragma unroll
      for (int h = 0; h < Hv; ++h) {
        v2bf pr;
        pr[0] = (__bf16)(temp * acch[h].x);
        pr[1] = (__bf16)(temp * acch[h].y);
        *(v2bf*)&QKbf[h * LPQ + l * PQ + s] = pr;
      }
    }
  }
  __syncthreads();

  // ---------------- Phase B2: softmax over S, all 768 (h,l) rows, in place ----------------
  for (int rid = wave; rid < Hv * Lv; rid += 8) {
    __bf16* row = &QKbf[rid * PQ];   // rid = h*96 + l  ->  h*LPQ + l*PQ
    float x[3];
#pragma unroll
    for (int j = 0; j < 3; ++j) x[j] = (float)row[lane + 32 * j];
    float mx = fmaxf(fmaxf(x[0], x[1]), x[2]);
#pragma unroll
    for (int off = 16; off >= 1; off >>= 1)
      mx = fmaxf(mx, __shfl_xor(mx, off, 32));
    float sum = 0.f;
#pragma unroll
    for (int j = 0; j < 3; ++j) { x[j] = __expf(x[j] - mx); sum += x[j]; }
#pragma unroll
    for (int off = 16; off >= 1; off >>= 1)
      sum += __shfl_xor(sum, off, 32);
    const float inv = 1.0f / sum;
#pragma unroll
    for (int j = 0; j < 3; ++j) row[lane + 32 * j] = (__bf16)(x[j] * inv);
  }

  // ---------------- Phase B3: V = A @ values, per head ----------------
  // wave w: dt = (w&3)*16 fixed, 3 lt strips -> B operand shared by 3 independent WMMAs
  const int dtI = (wave & 3) * 16;
  const int g   = wave >> 2;   // 0/1 -> lt strips {0..2} / {3..5}
  for (int h = 0; h < Hv; ++h) {
    __syncthreads();  // previous head's AV reads of bufA done; probs stable
    for (int i = t; i < Sv * Dv; i += 256) {  // stage V^T : bufA[d][s], row stride PQ
      int s = i >> 6, d = i & 63;
      bufA[d * PQ + s] = (__bf16)v[(((size_t)bn * Sv + s) * Hv + h) * Dv + d];
    }
    __syncthreads();

    const __bf16* Psh = &QKbf[h * LPQ];
    float* __restrict__ vp = v_out + (size_t)bn * (Lv * Hv * Dv) + h * Dv;
    const int lt0 = (g * 3 + 0) * 16, lt1 = (g * 3 + 1) * 16, lt2 = (g * 3 + 2) * 16;
    v8f acc0 = {}, acc1 = {}, acc2 = {};
#pragma unroll
    for (int sb = 0; sb < Sv; sb += 32) {
      const __bf16* br = &bufA[(dtI + mn) * PQ + sb + 8 * c];   // V^T row (dt+n)
      v16bf Bm = cat8(*(const v8bf*)br, *(const v8bf*)(br + 16));
      const __bf16* a0 = &Psh[(lt0 + mn) * PQ + sb + 8 * c];
      const __bf16* a1 = &Psh[(lt1 + mn) * PQ + sb + 8 * c];
      const __bf16* a2 = &Psh[(lt2 + mn) * PQ + sb + 8 * c];
      v16bf A0 = cat8(*(const v8bf*)a0, *(const v8bf*)(a0 + 16));
      v16bf A1 = cat8(*(const v8bf*)a1, *(const v8bf*)(a1 + 16));
      v16bf A2 = cat8(*(const v8bf*)a2, *(const v8bf*)(a2 + 16));
      acc0 = __builtin_amdgcn_wmma_f32_16x16x32_bf16(false, A0, false, Bm, (short)0, acc0, false, false);
      acc1 = __builtin_amdgcn_wmma_f32_16x16x32_bf16(false, A1, false, Bm, (short)0, acc1, false, false);
      acc2 = __builtin_amdgcn_wmma_f32_16x16x32_bf16(false, A2, false, Bm, (short)0, acc2, false, false);
    }
#pragma unroll
    for (int r = 0; r < 8; ++r) {
      const int d = dtI + mn;
      const int lr = r + 8 * c;
      vp[(lt0 + lr) * (Hv * Dv) + d] = acc0[r];
      vp[(lt1 + lr) * (Hv * Dv) + d] = acc1[r];
      vp[(lt2 + lr) * (Hv * Dv) + d] = acc2[r];
    }
  }
}

extern "C" void kernel_launch(void* const* d_in, const int* in_sizes, int n_in,
                              void* d_out, int out_size, void* d_ws, size_t ws_size,
                              hipStream_t stream) {
  const float* q    = (const float*)d_in[0];
  const float* k    = (const float*)d_in[1];
  const float* v    = (const float*)d_in[2];
  const float* mask = (const float*)d_in[3];
  const float* as   = (const float*)d_in[4];
  const float* W    = (const float*)d_in[5];
  const float* bv   = (const float*)d_in[6];
  // d_in[7], d_in[8]: query/key lengths (all full -> unused)

  float* qk_out = (float*)d_out;                         // [B,N,H,L,S]
  float* v_out  = qk_out + (size_t)BN_ * Hv * Lv * Sv;   // [B,N,L,H,D]

  stcat_attention_kernel<<<BN_, 256, LDS_BYTES, stream>>>(
      q, k, v, mask, as, W, bv, qk_out, v_out);
}
